// SpotlightScoreModel_89326729822791
// MI455X (gfx1250) — matrix operand order repaired and checked by reading, not compiled, not run on hardware
//
#include <hip/hip_runtime.h>
#include <hip/hip_fp16.h>
#include <math.h>

// Problem constants (from reference)
#define B_      8
#define D_      256
#define N_LIVE_ 10000
#define K_NEW_  32
#define KC_     10032          // N_LIVE + K_NEW  (multiple of 16)
#define KCP_    10048          // padded to multiple of 32 for WMMA K-tiling
#define NS_     100            // bootstrap samples
#define MROWS_  112            // 100 boot rows + 1 final-score row + 11 zero rows
#define EPS_    0.1f

typedef __attribute__((ext_vector_type(16))) _Float16 v16h;
typedef __attribute__((ext_vector_type(8)))  float    v8f;
typedef __attribute__((ext_vector_type(4)))  float    f32x4;

// Workspace layout (byte offsets, all 256B-aligned)
#define WS_ST   0ull            // _Float16 STt[B][D][KCP]    : scores^T, 41,156,608 B
#define WS_A    41156608ull     // _Float16 A  [B][MROWS][KCP]: weight rows, 18,006,016 B
#define WS_C    59162624ull     // float    C  [B][MROWS][D]  : GEMM out,     917,504 B
#define WS_LOG  60080128ull     // float    logits[B][KC]     :               321,024 B
// total ~60.4 MB

// ---------------------------------------------------------------------------
// Kernel 1: build S^T in f16.  S[b,k,d] = (x[b,k,d]-xt[b,d])/ts, stored
// transposed as STt[b][d][k] so the GEMM B-operand is k-contiguous per lane.
// LDS-tiled 16(k) x 64(d) transpose; pad column avoids bank conflicts.
// ---------------------------------------------------------------------------
__global__ void k_prep_scores(const float* __restrict__ xt,
                              const float* __restrict__ live_x0,
                              const float* __restrict__ x0,
                              const float* __restrict__ tsp,
                              _Float16* __restrict__ STt) {
  __shared__ float tile[16][65];
  const float inv_ts = 1.0f / tsp[0];
  const int k0 = blockIdx.x * 16;      // 627 tiles (N_LIVE boundary is tile-aligned)
  const int d0 = blockIdx.y * 64;      // 4 tiles
  const int b  = blockIdx.z;
  const int tid = threadIdx.x;         // 256 threads

  for (int e = tid; e < 16 * 64; e += 256) {
    const int lk = e >> 6, ld = e & 63;
    const int k = k0 + lk, d = d0 + ld;
    float x;
    if (k < N_LIVE_) x = live_x0[((size_t)b * N_LIVE_ + k) * D_ + d];
    else             x = x0[((size_t)b * K_NEW_ + (k - N_LIVE_)) * D_ + d];
    tile[lk][ld] = (x - xt[b * D_ + d]) * inv_ts;
  }
  __syncthreads();
  for (int e = tid; e < 16 * 64; e += 256) {
    const int lk = e & 15, ld = e >> 4;
    STt[((size_t)b * D_ + (d0 + ld)) * KCP_ + (k0 + lk)] = (_Float16)tile[lk][ld];
  }
}

// ---------------------------------------------------------------------------
// Kernel 2: logits[b,k].  For new particles the correction cancels -> ll.
// One wave per k; lane-strided squared-norm + shuffle reduce.
// ---------------------------------------------------------------------------
__global__ void k_logits(const float* __restrict__ xt,
                         const float* __restrict__ live_x0,
                         const float* __restrict__ live_ll,
                         const float* __restrict__ ll,
                         const float* __restrict__ tsp,
                         float* __restrict__ logits) {
  const int b    = blockIdx.y;
  const int wave = threadIdx.x >> 5;
  const int lane = threadIdx.x & 31;
  const int k    = blockIdx.x * 8 + wave;     // 1254*8 == 10032 exactly
  const float ts = tsp[0];
  float out;
  if (k >= N_LIVE_) {
    out = ll[b * K_NEW_ + (k - N_LIVE_)];
  } else {
    const float* xr = live_x0 + ((size_t)b * N_LIVE_ + k) * D_;
    const float* xb = xt + b * D_;
    float acc = 0.f;
#pragma unroll
    for (int i = 0; i < 8; ++i) {
      const int d = lane + 32 * i;
      const float df = xr[d] - xb[d];
      acc += df * df;
    }
    for (int m = 16; m >= 1; m >>= 1) acc += __shfl_xor(acc, m, 32);
    out = live_ll[b * N_LIVE_ + k] - 0.5f * acc / (ts * ts) - logf(ts);
  }
  if (lane == 0) logits[(size_t)b * KC_ + k] = out;
}

// ---------------------------------------------------------------------------
// Kernel 3: build one GEMM A-row per block.
//   m < 100 : bootstrap softmax over gathered logits, scatter-add into LDS row
//   m == 100: plain softmax over all logits (final score weights)
//   m > 100 : zero pad row
// Row (KCP floats = 40KB) lives entirely in LDS -> no global atomics.
// ---------------------------------------------------------------------------
__global__ void k_weights(const float* __restrict__ logits,
                          const int*   __restrict__ boot_idx,
                          _Float16*    __restrict__ A) {
  __shared__ float row[KCP_];
  __shared__ float red[8];
  const int m = blockIdx.x;          // 0..111
  const int b = blockIdx.y;
  const int tid = threadIdx.x;       // 256
  const int wave = tid >> 5, lane = tid & 31;

  for (int k = tid; k < KCP_; k += 256) row[k] = 0.f;
  __syncthreads();

  const float* lb = logits + (size_t)b * KC_;

  if (m <= NS_) {
    const int* ib = (m < NS_) ? (boot_idx + ((size_t)m * B_ + b) * KC_) : nullptr;

    // pass 1: max of (gathered) logits
    float lmax = -3.4e38f;
    for (int k = tid; k < KC_; k += 256) {
      const int j = ib ? ib[k] : k;
      lmax = fmaxf(lmax, lb[j]);
    }
    for (int s = 16; s >= 1; s >>= 1) lmax = fmaxf(lmax, __shfl_xor(lmax, s, 32));
    if (lane == 0) red[wave] = lmax;
    __syncthreads();
    if (tid == 0) {
      float v = red[0];
      for (int i = 1; i < 8; ++i) v = fmaxf(v, red[i]);
      red[0] = v;
    }
    __syncthreads();
    lmax = red[0];
    __syncthreads();

    // pass 2: sum of exp
    float se = 0.f;
    for (int k = tid; k < KC_; k += 256) {
      const int j = ib ? ib[k] : k;
      se += __expf(lb[j] - lmax);
    }
    for (int s = 16; s >= 1; s >>= 1) se += __shfl_xor(se, s, 32);
    if (lane == 0) red[wave] = se;
    __syncthreads();
    if (tid == 0) {
      float v = 0.f;
      for (int i = 0; i < 8; ++i) v += red[i];
      red[0] = v;
    }
    __syncthreads();
    const float invZ = 1.0f / red[0];

    // pass 3: scatter (bootstrap) or direct fill (final)
    if (m < NS_) {
      for (int k = tid; k < KC_; k += 256) {
        const int j = ib[k];
        atomicAdd(&row[j], __expf(lb[j] - lmax) * invZ);
      }
    } else {
      for (int k = tid; k < KC_; k += 256)
        row[k] = __expf(lb[k] - lmax) * invZ;
    }
  }
  __syncthreads();

  _Float16* ar = A + ((size_t)b * MROWS_ + m) * KCP_;
  for (int k = tid; k < KCP_; k += 256) ar[k] = (_Float16)row[k];
}

// ---------------------------------------------------------------------------
// Kernel 4: WMMA GEMM.  C[b] (112x256) = A[b] (112xKCP, f16) @ S[b] (KCPx256).
// One wave per 16x16 output tile; K-loop of KCP/32 = 314 v_wmma_f32_16x16x32_f16.
// A lane layout (16x32 f16, per ISA 7.12.2): lane r<16 holds row r, K chunks
// {0..7, 16..23}; lane r+16 holds K chunks {8..15, 24..31}.
// B lane layout (32x16): lane n<16 holds col n K {0..7, 16..23}; lane n+16 K {8..15, 24..31}.
// Both are contiguous 16B loads thanks to the S^T layout. No divergence ->
// EXEC is all-ones at every WMMA (ISA requirement).
// ---------------------------------------------------------------------------
__global__ void k_gemm(const _Float16* __restrict__ STt,
                       const _Float16* __restrict__ A,
                       float* __restrict__ C) {
  const int w    = (blockIdx.x << 2) + (threadIdx.x >> 5);   // 0..895
  const int lane = threadIdx.x & 31;
  const int b    = w / 112;
  const int t    = w % 112;
  const int m0   = (t >> 4) << 4;
  const int n0   = (t & 15) << 4;
  const int r    = lane & 15;
  const int hi   = lane >> 4;

  const _Float16* ap = A   + ((size_t)b * MROWS_ + (m0 + r)) * KCP_ + hi * 8;
  const _Float16* bp = STt + ((size_t)b * D_     + (n0 + r)) * KCP_ + hi * 8;

  union V { v16h v; f32x4 q[2]; };
  v8f acc = {};
#pragma unroll 2
  for (int kk = 0; kk < KCP_; kk += 32) {
    V av, bv;
    av.q[0] = *(const f32x4*)(ap + kk);        // K {0..7} / {8..15}
    av.q[1] = *(const f32x4*)(ap + kk + 16);   // K {16..23} / {24..31}
    bv.q[0] = *(const f32x4*)(bp + kk);        // K {0..7} / {8..15}
    bv.q[1] = *(const f32x4*)(bp + kk + 16);   // K {16..23} / {24..31}
    acc = __builtin_amdgcn_wmma_f32_16x16x32_f16(
        /*neg_a=*/false, av.v, /*neg_b=*/false, bv.v,
        /*c_mod=*/(short)0, acc, /*reuse_a=*/false, /*reuse_b=*/false);
  }

  // C/D layout: VGPR j, lanes 0-15 -> M=m0+j, N=lane; lanes 16-31 -> M=m0+j+8
  float* cb = C + (size_t)b * MROWS_ * D_;
#pragma unroll
  for (int j = 0; j < 8; ++j)
    cb[(size_t)(m0 + j + 8 * hi) * D_ + (n0 + r)] = acc[j];
}

// ---------------------------------------------------------------------------
// Kernel 5: unbiased std over boot rows 0..99 per (b,d), convergence flag,
// and emit score (row 100).  out = [score (B*D floats) | converged (B floats)]
// ---------------------------------------------------------------------------
__global__ void k_final(const float* __restrict__ C,
                        const float* __restrict__ tsp,
                        float* __restrict__ out) {
  __shared__ int ok;
  const int b = blockIdx.x, d = threadIdx.x;   // 256 threads
  if (d == 0) ok = 1;
  __syncthreads();
  const float* cb = C + (size_t)b * MROWS_ * D_;
  float s = 0.f, s2 = 0.f;
  for (int n = 0; n < NS_; ++n) {
    const float v = cb[(size_t)n * D_ + d];
    s += v; s2 += v * v;
  }
  const float mean  = s * (1.0f / NS_);
  const float var   = (s2 - (float)NS_ * mean * mean) * (1.0f / (NS_ - 1));
  const float sigma = sqrtf(fmaxf(var, 0.f));
  if (!(sigma < EPS_ * tsp[0])) atomicAnd(&ok, 0);
  out[b * D_ + d] = cb[(size_t)NS_ * D_ + d];
  __syncthreads();
  if (d == 0) out[B_ * D_ + b] = ok ? 1.0f : 0.0f;
}

// ---------------------------------------------------------------------------
extern "C" void kernel_launch(void* const* d_in, const int* in_sizes, int n_in,
                              void* d_out, int out_size, void* d_ws, size_t ws_size,
                              hipStream_t stream) {
  const float* xt      = (const float*)d_in[0];
  const float* live_x0 = (const float*)d_in[1];
  const float* live_ll = (const float*)d_in[2];
  const float* x0      = (const float*)d_in[3];
  const float* ll      = (const float*)d_in[4];
  const float* ts      = (const float*)d_in[5];
  const int*   bidx    = (const int*)d_in[6];
  float* out = (float*)d_out;

  char* ws = (char*)d_ws;
  _Float16* STt   = (_Float16*)(ws + WS_ST);
  _Float16* A     = (_Float16*)(ws + WS_A);
  float*    C     = (float*)(ws + WS_C);
  float*    logit = (float*)(ws + WS_LOG);

  k_prep_scores<<<dim3(627, 4, 8), 256, 0, stream>>>(xt, live_x0, x0, ts, STt);
  k_logits     <<<dim3(1254, 8),   256, 0, stream>>>(xt, live_x0, live_ll, ll, ts, logit);
  k_weights    <<<dim3(112, 8),    256, 0, stream>>>(logit, bidx, A);
  k_gemm       <<<224,             128, 0, stream>>>(STt, A, C);
  k_final      <<<8,               256, 0, stream>>>(C, ts, out);
}
